// BiRecurrent_26439818674416
// MI455X (gfx1250) — compile-verified, matrix operand order checked by reading, not run on hardware
//
#include <hip/hip_runtime.h>
#include <hip/hip_bf16.h>

// Bidirectional 2-layer LSTM for MI455X (gfx1250), wave32 + WMMA bf16.
//
//   1. Convert weights/inputs fp32->bf16 (bf16 WMMA, fp32 accumulate).
//   2. Z0[s,b,4H] = x @ Wih0^T + bih0 + bhh0                      (WMMA)
//   3. Persistent recurrence, layer 0, fwd+rev fused: 512 steps of
//      32x512x2048 GEMM via v_wmma_f32_16x16x32_bf16 + grid barrier.
//   4. Z1f/Z1r from layer-0 hidden streams.                        (WMMA)
//   5. Persistent recurrence, layer 1, fwd+rev fused.
//   6. Concat(+time-unreverse) -> output GEMM out @ Wout^T.        (WMMA)
//   7. Tiny hid/cel projections.

#define S_LEN 512
#define BATCH 32
#define DIMD  512
#define HID   512
#define NGATE 2048  // 4*HID
#define LDSROW 528  // padded LDS row stride (bf16 elems), 32B aligned

typedef __attribute__((ext_vector_type(16))) __bf16 v16bf;
typedef __attribute__((ext_vector_type(8)))  float  v8f;

// ---------------------------------------------------------------------------
// Branch-free activations on hardware transcendentals (v_tanh_f32 / v_exp_f32
// / v_rcp_f32 on gfx1250). Guarded so either toolchain compiles.
// ---------------------------------------------------------------------------
__device__ __forceinline__ float fast_exp2(float x) {
#if __has_builtin(__builtin_amdgcn_exp2f)
  return __builtin_amdgcn_exp2f(x);
#else
  return exp2f(x);
#endif
}
__device__ __forceinline__ float fast_rcp(float x) {
#if __has_builtin(__builtin_amdgcn_rcpf)
  return __builtin_amdgcn_rcpf(x);
#else
  return 1.0f / x;
#endif
}
__device__ __forceinline__ float sigmoidf_(float x) {
  // 1 / (1 + 2^(-x*log2(e)))
  return fast_rcp(1.0f + fast_exp2(-1.442695041f * x));
}
__device__ __forceinline__ float tanhf_(float x) {
#if __has_builtin(__builtin_amdgcn_tanhf)
  return __builtin_amdgcn_tanhf(x);
#else
  // tanh(x) = 1 - 2 / (1 + 2^(2*x*log2(e)))
  return 1.0f - 2.0f * fast_rcp(1.0f + fast_exp2(2.885390082f * x));
#endif
}

// ---------------------------------------------------------------------------
// bf16 WMMA GEMM: C[M,N](f32) = A[M,K](bf16) * W[N,K]^T(bf16) + bias0(+bias1)
// MODE 0: row-major C[m*N+n]   (Z buffers, step-major rows m = s*B+b)
// MODE 1: final output remap   out[(b*S + s)*N + n],  b=m%32, s=m/32
//         (m0 is a 32-multiple, so s = m0>>5 is wave-uniform: cheap remap)
// Block = 128 threads = 4 waves; block tile 64x64; wave tile 32x32 (4 accums).
// ---------------------------------------------------------------------------
template <int MODE, bool HAS_B1>
__global__ void __launch_bounds__(128)
wmma_gemm_bf16(const __bf16* __restrict__ A, const __bf16* __restrict__ W,
               float* __restrict__ C, const float* __restrict__ bias0,
               const float* __restrict__ bias1, int M, int N, int K) {
  const int lane = threadIdx.x & 31;
  const int wave = threadIdx.x >> 5;
  const int wm = wave & 1, wn = wave >> 1;
  const int m0 = blockIdx.x * 64 + wm * 32;
  const int n0 = blockIdx.y * 64 + wn * 32;
  const int ar = lane & 15;         // tile row (M for A-frag, N for B-frag)
  const int kq = (lane >> 4) * 16;  // which 16-element K half of the 32-chunk
  const int hb = lane >> 4;
  const int nn = lane & 15;

  // Bias is a function of n only; hoist (2 scalar loads, overlap with K loop).
  float bv[2];
#pragma unroll
  for (int j = 0; j < 2; ++j) {
    int n = n0 + 16 * j + nn;
    bv[j] = bias0[n];
    if (HAS_B1) bv[j] += bias1[n];
  }

  v8f acc[2][2] = {};
  for (int k = 0; k < K; k += 32) {
    v16bf a[2], b[2];
#pragma unroll
    for (int i = 0; i < 2; ++i)
      a[i] = *reinterpret_cast<const v16bf*>(A + (size_t)(m0 + 16 * i + ar) * K + k + kq);
#pragma unroll
    for (int j = 0; j < 2; ++j)
      b[j] = *reinterpret_cast<const v16bf*>(W + (size_t)(n0 + 16 * j + ar) * K + k + kq);
#pragma unroll
    for (int i = 0; i < 2; ++i)
#pragma unroll
      for (int j = 0; j < 2; ++j)
        acc[i][j] = __builtin_amdgcn_wmma_f32_16x16x32_bf16(
            false, a[i], false, b[j], (short)0, acc[i][j], false, false);
  }

  // Branch-free epilogue. C layout per VGPR g: m = g + 8*(lane>>4), n = lane&15.
#pragma unroll
  for (int i = 0; i < 2; ++i) {
    if (MODE == 0) {
      size_t base = (size_t)(m0 + 16 * i + 8 * hb) * N + n0 + nn;
#pragma unroll
      for (int j = 0; j < 2; ++j)
#pragma unroll
        for (int g = 0; g < 8; ++g)
          C[base + (size_t)g * N + 16 * j] = acc[i][j][g] + bv[j];
    } else {
      const int ss = m0 >> 5;  // wave-uniform sequence index
      const int mb = 16 * i + 8 * hb;
      size_t base = (size_t)ss * N + n0 + nn;
#pragma unroll
      for (int j = 0; j < 2; ++j)
#pragma unroll
        for (int g = 0; g < 8; ++g)
          C[base + (size_t)(mb + g) * S_LEN * N + 16 * j] = acc[i][j][g] + bv[j];
    }
  }
}

// ---------------------------------------------------------------------------
// fp32 -> bf16 conversion (weights)
// ---------------------------------------------------------------------------
__global__ void cvt_f32_bf16(const float* __restrict__ src,
                             __bf16* __restrict__ dst, size_t n) {
  size_t i = (size_t)blockIdx.x * blockDim.x + threadIdx.x;
  size_t stride = (size_t)gridDim.x * blockDim.x;
  for (; i < n; i += stride) dst[i] = (__bf16)src[i];
}

// X:(B,S,D) fp32 -> Xb:[(s*B+b), D] bf16  (step-major rows for the Z0 GEMM)
__global__ void cvt_x_transpose(const float* __restrict__ X,
                                __bf16* __restrict__ Xb) {
  size_t total = (size_t)BATCH * S_LEN * DIMD;
  size_t stride = (size_t)gridDim.x * blockDim.x;
  for (size_t i = (size_t)blockIdx.x * blockDim.x + threadIdx.x; i < total; i += stride) {
    size_t k = i % DIMD;
    size_t rest = i / DIMD;
    size_t s = rest % S_LEN;
    size_t b = rest / S_LEN;
    Xb[((s * BATCH) + b) * DIMD + k] = (__bf16)X[i];
  }
}

// Concat layer-1 hidden streams (un-reversing H1r in time) for output GEMM.
// Cc[(s*B+b), 0:512] = H1f[s,b,:] ; Cc[(s*B+b), 512:1024] = H1r[S-1-s, b, :]
__global__ void concat_out(const __bf16* __restrict__ Hf,
                           const __bf16* __restrict__ Hr,
                           __bf16* __restrict__ Cc) {
  size_t total = (size_t)S_LEN * BATCH * 2 * HID;
  size_t stride = (size_t)gridDim.x * blockDim.x;
  for (size_t i = (size_t)blockIdx.x * blockDim.x + threadIdx.x; i < total; i += stride) {
    size_t k = i % (2 * HID);
    size_t row = i / (2 * HID);  // s*B + b
    size_t b = row % BATCH, s = row / BATCH;
    __bf16 v;
    if (k < HID)
      v = Hf[(s * BATCH + b) * HID + k];
    else
      v = Hr[(((size_t)(S_LEN - 1 - s)) * BATCH + b) * HID + (k - HID)];
    Cc[i] = v;
  }
}

// ---------------------------------------------------------------------------
// Persistent LSTM recurrence, fwd+rev fused.
// 128 blocks x 128 threads. Blocks 0-63: dir 0 (fwd), 64-127: dir 1 (rev).
// Each block owns 8 hidden columns (jb..jb+7) for all 32 batches:
//   gates[32b x 32rows] = h[32 x 512] @ Whh_slice^T   (WMMA bf16, LDS weights)
// then fused activations + c/h update, bf16 h broadcast via global state
// buffer, per-step grid barrier (monotonic atomic counter + s_sleep spin).
// ---------------------------------------------------------------------------
__global__ void __launch_bounds__(128)
lstm_recurrence(const float* __restrict__ Zd0, const float* __restrict__ Zd1,
                const float* __restrict__ Whh,   // [4H][H] fp32 (layer slice)
                const float* __restrict__ h0,    // [B][H] fp32 (layer slice)
                const float* __restrict__ c0,    // [B][H] fp32
                __bf16* __restrict__ hstate0, __bf16* __restrict__ hstate1,
                __bf16* __restrict__ Hout0, __bf16* __restrict__ Hout1,
                float* __restrict__ hfin,        // base of [2][B][H]; +dir*B*H
                float* __restrict__ cfin,
                unsigned* __restrict__ sync0, unsigned* __restrict__ sync1,
                int zrev) {
  const int dir = blockIdx.x >> 6;
  const int wg = blockIdx.x & 63;
  const int jb = wg * 8;

  const float* Z = dir ? Zd1 : Zd0;
  __bf16* hstate = dir ? hstate1 : hstate0;
  __bf16* Hout = dir ? Hout1 : Hout0;
  unsigned* sync = dir ? sync1 : sync0;
  float* hfinp = hfin + (size_t)dir * BATCH * HID;
  float* cfinp = cfin + (size_t)dir * BATCH * HID;

  const int tid = threadIdx.x;
  const int lane = tid & 31;
  const int wave = tid >> 5;
  const int msub = wave & 1;   // which 16 batches
  const int nsub = wave >> 1;  // which 16 gate-rows of the 32-row set

  __shared__ __attribute__((aligned(32))) __bf16 sW[32 * LDSROW];
  __shared__ float sG[32 * 32];

  // Preload + convert this block's 32 Whh rows (i,f,g,o for cols jb..jb+7).
  for (int idx = tid; idx < 32 * 512; idx += 128) {
    int r32 = idx >> 9;  // 0..31
    int k = idx & 511;
    int r = (r32 >> 3) * HID + jb + (r32 & 7);  // actual Whh row
    sW[r32 * LDSROW + k] = (__bf16)Whh[(size_t)r * HID + k];
  }

  // Init owned c/h slice (2 (batch,col) pairs per thread).
  float creg[2], hreg[2];
#pragma unroll
  for (int e = 0; e < 2; ++e) {
    int idx = tid * 2 + e;
    int b = idx >> 3, col = jb + (idx & 7);
    creg[e] = c0[(size_t)b * HID + col];
    hreg[e] = h0[(size_t)b * HID + col];
    hstate[(size_t)b * HID + col] = (__bf16)hreg[e];
  }
  __syncthreads();

  unsigned epoch = 0;
  auto gsync = [&]() {
    __syncthreads();
    ++epoch;
    if (tid == 0) {
      __threadfence();  // release our h writes to device scope
      atomicAdd(sync, 1u);
      while (atomicAdd(sync, 0u) < epoch * 64u) __builtin_amdgcn_s_sleep(1);
    }
    __syncthreads();
    __threadfence();  // acquire other blocks' h writes
  };

  gsync();  // make h0 visible everywhere before step 0

  const int ar = lane & 15;
  const int kq = (lane >> 4) * 16;
  const int r32 = 16 * nsub + ar;  // B-fragment row within the 32-row set

  for (int t = 0; t < S_LEN; ++t) {
    const int tz = (dir && zrev) ? (S_LEN - 1 - t) : t;
    const float* Zt = Z + (size_t)tz * BATCH * NGATE;

    if (t + 1 < S_LEN) {  // warm L2/L0 for next step's gate biases
      int tz2 = (dir && zrev) ? (S_LEN - 2 - t) : (t + 1);
      __builtin_prefetch(Z + (size_t)tz2 * BATCH * NGATE + (size_t)tid * 64, 0, 1);
    }

    // gates(32x32) = h(32x512,bf16) @ WhhSlice^T : 16 WMMAs per wave.
    v8f acc = {};
#pragma unroll 4
    for (int k = 0; k < HID; k += 32) {
      v16bf a = *reinterpret_cast<const v16bf*>(
          hstate + (size_t)(16 * msub + ar) * HID + k + kq);
      v16bf b = *reinterpret_cast<const v16bf*>(&sW[r32 * LDSROW + k + kq]);
      acc = __builtin_amdgcn_wmma_f32_16x16x32_bf16(false, a, false, b,
                                                    (short)0, acc, false, false);
    }
#pragma unroll
    for (int g = 0; g < 8; ++g) {
      int m = 16 * msub + g + 8 * (lane >> 4);  // batch
      int n = 16 * nsub + (lane & 15);          // gate-row-in-set
      sG[m * 32 + n] = acc[g];
    }
    __syncthreads();

    // Fused activations + state update for owned (batch,col) pairs.
#pragma unroll
    for (int e = 0; e < 2; ++e) {
      int idx = tid * 2 + e;
      int b = idx >> 3, ci = idx & 7, col = jb + ci;
      const float* zrow = Zt + (size_t)b * NGATE;
      float gi = sG[b * 32 + ci]      + zrow[col];
      float gf = sG[b * 32 + 8 + ci]  + zrow[HID + col];
      float gg = sG[b * 32 + 16 + ci] + zrow[2 * HID + col];
      float go = sG[b * 32 + 24 + ci] + zrow[3 * HID + col];
      float cn = sigmoidf_(gf) * creg[e] + sigmoidf_(gi) * tanhf_(gg);
      float hn = sigmoidf_(go) * tanhf_(cn);
      creg[e] = cn;
      hreg[e] = hn;
      __bf16 hb16 = (__bf16)hn;
      hstate[(size_t)b * HID + col] = hb16;
      Hout[((size_t)t * BATCH + b) * HID + col] = hb16;
    }
    gsync();
  }

  // Final states (fp32) for hid/cel projections.
#pragma unroll
  for (int e = 0; e < 2; ++e) {
    int idx = tid * 2 + e;
    int b = idx >> 3, col = jb + (idx & 7);
    hfinp[(size_t)b * HID + col] = hreg[e];
    cfinp[(size_t)b * HID + col] = creg[e];
  }
}

// ---------------------------------------------------------------------------
// hid = concat(hT_f, hT_r) @ Whid^T + bhid ; cel likewise. 64x1024x512 each —
// trivial FLOPs, simple fp32 dot kernel.
// ---------------------------------------------------------------------------
__global__ void hidcel_kernel(const float* __restrict__ hfin,
                              const float* __restrict__ cfin,
                              const float* __restrict__ Whid,
                              const float* __restrict__ bhid,
                              const float* __restrict__ Wcell,
                              const float* __restrict__ bcell,
                              float* __restrict__ out_hid,
                              float* __restrict__ out_cel) {
  int idx = blockIdx.x * blockDim.x + threadIdx.x;  // 0..65535
  int which = idx >> 15;  // 0 = hid, 1 = cel
  int rem = idx & 32767;
  int row = rem >> 9;  // l*32 + b
  int n = rem & 511;
  int l = row >> 5, b = row & 31;
  const float* st = which ? cfin : hfin;
  const float* W = which ? Wcell : Whid;
  const float* bias = which ? bcell : bhid;
  const float* af = st + ((size_t)(2 * l) * BATCH + b) * HID;       // fwd pass
  const float* arv = st + ((size_t)(2 * l + 1) * BATCH + b) * HID;  // rev pass
  const float* wrow = W + (size_t)n * (2 * HID);
  float acc = bias[n];
  for (int k = 0; k < HID; ++k) acc += af[k] * wrow[k];
  for (int k = 0; k < HID; ++k) acc += arv[k] * wrow[HID + k];
  float* out = which ? out_cel : out_hid;
  out[(size_t)row * 512 + n] = acc;
}

// ---------------------------------------------------------------------------
extern "C" void kernel_launch(void* const* d_in, const int* in_sizes, int n_in,
                              void* d_out, int out_size, void* d_ws,
                              size_t ws_size, hipStream_t stream) {
  const float* X     = (const float*)d_in[0];   // (B,S,D)
  const float* h0    = (const float*)d_in[1];   // (L,B,H)
  const float* c0    = (const float*)d_in[2];
  const float* Wih   = (const float*)d_in[3];   // (L,4H,D)
  const float* Whh   = (const float*)d_in[4];   // (L,4H,H)
  const float* bih   = (const float*)d_in[5];   // (L,4H)
  const float* bhh   = (const float*)d_in[6];
  const float* Wout  = (const float*)d_in[7];   // (512,2H)
  const float* bout  = (const float*)d_in[8];
  const float* Whid  = (const float*)d_in[9];
  const float* bhid  = (const float*)d_in[10];
  const float* Wcell = (const float*)d_in[11];
  const float* bcell = (const float*)d_in[12];
  float* out = (float*)d_out;

  char* ws = (char*)d_ws;
  size_t off = 0;
  auto walloc = [&](size_t bytes) -> void* {
    void* p = ws + off;
    off = (off + bytes + 255) & ~(size_t)255;
    return p;
  };

  const size_t ZBYTES = (size_t)S_LEN * BATCH * NGATE * sizeof(float);  // 128MB
  const size_t HBYTES = (size_t)S_LEN * BATCH * HID * sizeof(__bf16);   // 16MB

  float*  Zf    = (float*)walloc(ZBYTES);
  float*  Zr    = (float*)walloc(ZBYTES);
  __bf16* Xb    = (__bf16*)walloc((size_t)S_LEN * BATCH * DIMD * 2);
  __bf16* Wih0b = (__bf16*)walloc((size_t)NGATE * DIMD * 2);
  __bf16* Wih1b = (__bf16*)walloc((size_t)NGATE * HID * 2);
  __bf16* Woutb = (__bf16*)walloc((size_t)512 * 2 * HID * 2);
  __bf16* H0f   = (__bf16*)walloc(HBYTES);
  __bf16* H0r   = (__bf16*)walloc(HBYTES);
  __bf16* H1f   = (__bf16*)walloc(HBYTES);
  __bf16* H1r   = (__bf16*)walloc(HBYTES);
  __bf16* Cc    = (__bf16*)walloc((size_t)S_LEN * BATCH * 2 * HID * 2);
  __bf16* hs0   = (__bf16*)walloc((size_t)BATCH * HID * 2);
  __bf16* hs1   = (__bf16*)walloc((size_t)BATCH * HID * 2);
  float*  hfin  = (float*)walloc((size_t)4 * BATCH * HID * sizeof(float));
  float*  cfin  = (float*)walloc((size_t)4 * BATCH * HID * sizeof(float));
  unsigned* syncc = (unsigned*)walloc(256);

  hipMemsetAsync(syncc, 0, 256, stream);

  // fp32 -> bf16 weight/input conversions.
  cvt_f32_bf16<<<1024, 256, 0, stream>>>(Wih, Wih0b, (size_t)NGATE * DIMD);
  cvt_f32_bf16<<<1024, 256, 0, stream>>>(Wih + (size_t)NGATE * DIMD, Wih1b,
                                         (size_t)NGATE * HID);
  cvt_f32_bf16<<<1024, 256, 0, stream>>>(Wout, Woutb, (size_t)512 * 2 * HID);
  cvt_x_transpose<<<4096, 256, 0, stream>>>(X, Xb);

  dim3 gz(256, 32);  // M=16384/64, N=2048/64

  // Layer 0 input projection (shared by both directions).
  wmma_gemm_bf16<0, true><<<gz, 128, 0, stream>>>(
      Xb, Wih0b, Zf, bih, bhh, S_LEN * BATCH, NGATE, DIMD);
  // Layer 0 recurrence, fwd + rev fused (rev reads Z0 time-reversed).
  lstm_recurrence<<<128, 128, 0, stream>>>(
      Zf, Zf, Whh, h0, c0, hs0, hs1, H0f, H0r, hfin, cfin,
      syncc + 0, syncc + 16, /*zrev=*/1);

  // Layer 1 input projections (step-indexed inputs -> no time reversal).
  wmma_gemm_bf16<0, true><<<gz, 128, 0, stream>>>(
      H0f, Wih1b, Zf, bih + NGATE, bhh + NGATE, S_LEN * BATCH, NGATE, HID);
  wmma_gemm_bf16<0, true><<<gz, 128, 0, stream>>>(
      H0r, Wih1b, Zr, bih + NGATE, bhh + NGATE, S_LEN * BATCH, NGATE, HID);
  lstm_recurrence<<<128, 128, 0, stream>>>(
      Zf, Zr, Whh + (size_t)NGATE * HID, h0 + (size_t)BATCH * HID,
      c0 + (size_t)BATCH * HID, hs0, hs1, H1f, H1r,
      hfin + (size_t)2 * BATCH * HID, cfin + (size_t)2 * BATCH * HID,
      syncc + 32, syncc + 48, /*zrev=*/0);

  // Output projection: concat(H1f, un-reversed H1r) @ Wout^T + bout.
  concat_out<<<4096, 256, 0, stream>>>(H1f, H1r, Cc);
  dim3 go(256, 8);  // M=16384/64, N=512/64
  wmma_gemm_bf16<1, false><<<go, 128, 0, stream>>>(
      Cc, Woutb, out, bout, nullptr, S_LEN * BATCH, 512, 2 * HID);

  // hid / cel projections from final states.
  float* out_hid = out + (size_t)BATCH * S_LEN * 512;  // 8388608
  float* out_cel = out_hid + (size_t)2 * BATCH * 512;  // +32768
  hidcel_kernel<<<256, 256, 0, stream>>>(hfin, cfin, Whid, bhid, Wcell, bcell,
                                         out_hid, out_cel);
}
